// Model_64209761075499
// MI455X (gfx1250) — matrix operand order repaired and checked by reading, not compiled, not run on hardware
//
#include <hip/hip_runtime.h>
#include <hip/hip_bf16.h>
#include <math.h>
#include <stdint.h>

typedef __bf16 bf16_t;
typedef __attribute__((ext_vector_type(16))) __bf16 v16bf;
typedef __attribute__((ext_vector_type(8)))  __bf16 v8bf;
typedef __attribute__((ext_vector_type(8)))  float  v8f;

typedef unsigned int tdm_v4u __attribute__((ext_vector_type(4)));
typedef int          tdm_v4i __attribute__((ext_vector_type(4)));
typedef int          tdm_v8i __attribute__((ext_vector_type(8)));

namespace {

constexpr int kDim  = 300;
constexpr int kDP   = 320;          // DIM padded to multiple of 32
constexpr int kNC   = 256;
constexpr int kKS   = 3;
constexpr int kNCls = 3;
constexpr int kB    = 128;
constexpr int kL1   = 512;
constexpr int kL2   = 32;
constexpr int kRows = kL1 + 2;      // halo row above & below
constexpr int kGID  = 2 * kDP;      // gate-input padded width (640)
constexpr int kK1   = kKS * kDP;    // 960  (sentiment conv K)
constexpr int kK2   = kKS * kGID;   // 1920 (gate conv K)
constexpr int kConvRowsPerBlk = 128;            // output rows per block
constexpr int kConvLdsRows    = kConvRowsPerBlk + 2;  // +halo
constexpr int kConvLdsBytes   = kConvLdsRows * kGID * 2;  // 166400 B (<320KB WGP LDS)

// ---------------- WMMA fragment helpers (A/B in contiguous k order; C/D per CDNA5 ISA 7.12.2) ----

// A fragment, 16x32 bf16, row-major source. Lane m(<16): K k..k+15;
// lane m+16: K k+16..k+31.
__device__ inline v16bf load_a(const bf16_t* A, int lda, int m, int k, int lane) {
  const bf16_t* p = A + (size_t)(m + (lane & 15)) * lda + k + ((lane & 16) ? 16 : 0);
  v8bf lo = *(const v8bf*)p;          // 16B
  v8bf hi = *(const v8bf*)(p + 8);    // 16B
  v16bf r;
#pragma unroll
  for (int i = 0; i < 8; ++i) { r[i] = lo[i]; r[i + 8] = hi[i]; }
  return r;
}

// B fragment, 32x16 bf16, from an N-major (transposed) matrix BT[n,k].
// Lane n(<16): K k..k+15; lane n+16: K k+16..k+31 -> one contiguous 32B load.
__device__ inline v16bf load_b(const bf16_t* BT, int ldb, int n, int k, int lane) {
  const bf16_t* p = BT + (size_t)(n + (lane & 15)) * ldb + k + ((lane & 16) ? 16 : 0);
  return *(const v16bf*)p;
}

__device__ inline v8f wmma_bf16(v16bf a, v16bf b, v8f c) {
  return __builtin_amdgcn_wmma_f32_16x16x32_bf16(
      /*neg_a=*/false, a, /*neg_b=*/false, b,
      /*c_mod=*/(short)0, c, /*reuse_a=*/false, /*reuse_b=*/false);
}

// C/D layout: lane n(<16) holds rows m..m+7 col n; lane n+16 rows m+8..m+15.
__device__ inline void store_f32(float* C, int ldc, int m, int n, int lane, v8f c) {
  float* p = C + (size_t)(m + ((lane & 16) ? 8 : 0)) * ldc + n + (lane & 15);
#pragma unroll
  for (int r = 0; r < 8; ++r) p[(size_t)r * ldc] = c[r];
}

__device__ inline void store_bf16(bf16_t* C, int ldc, int m, int n, int lane, v8f c,
                                  bool relu) {
  bf16_t* p = C + (size_t)(m + ((lane & 16) ? 8 : 0)) * ldc + n + (lane & 15);
#pragma unroll
  for (int r = 0; r < 8; ++r) {
    float v = c[r];
    if (relu && v < 0.f) v = 0.f;
    p[(size_t)r * ldc] = (bf16_t)v;
  }
}

// (MT*16) x (NT*16) per-wave GEMM, K multiple of 32. B frags reused across all
// M tiles, A frags across all N tiles -> 1 b128 load per WMMA at MT=NT=4.
template <int MT, int NT>
__device__ inline void wave_gemm(const bf16_t* A, int lda,
                                 const bf16_t* BT, int ldb, int K,
                                 int m_base, int n_base, int lane,
                                 v8f (&acc)[MT][NT]) {
#pragma unroll
  for (int i = 0; i < MT; ++i)
#pragma unroll
    for (int j = 0; j < NT; ++j) acc[i][j] = (v8f){0, 0, 0, 0, 0, 0, 0, 0};
  for (int k = 0; k < K; k += 32) {
    __builtin_prefetch(A + (size_t)(m_base + (lane & 15)) * lda + k + 64, 0, 3);
    v16bf bf[NT];
#pragma unroll
    for (int j = 0; j < NT; ++j) bf[j] = load_b(BT, ldb, n_base + j * 16, k, lane);
#pragma unroll
    for (int i = 0; i < MT; ++i) {
      v16bf a = load_a(A, lda, m_base + i * 16, k, lane);
#pragma unroll
      for (int j = 0; j < NT; ++j) acc[i][j] = wmma_bf16(a, bf[j], acc[i][j]);
    }
  }
}

__device__ inline void atomicMaxF(float* addr, float val) {
  int* ai = (int*)addr;
  int old = *ai;
  while (val > __int_as_float(old)) {
    int assumed = old;
    old = atomicCAS(ai, assumed, __float_as_int(val));
    if (old == assumed) break;
  }
}

// TDM: DMA a 2D bf16 tile (rows x width, row stride in elements) from global
// memory into LDS at byte offset lds_off. D# packing per CDNA5 ISA ch.8.
__device__ inline void tdm_load_2d_to_lds(const void* gptr, unsigned lds_off,
                                          unsigned width_elems, unsigned rows,
                                          unsigned stride_elems) {
  unsigned long long ga = (unsigned long long)(uintptr_t)gptr;
  tdm_v4u g0;
  g0[0] = 1u;                                   // count=1, user descriptor
  g0[1] = lds_off;                              // lds_addr (bytes)
  g0[2] = (unsigned)ga;                         // global_addr[31:0]
  g0[3] = (unsigned)((ga >> 32) & 0x1FFFFFFu) | (2u << 30);  // addr[56:32], type=2
  tdm_v8i g1;
  g1[0] = (int)(1u << 16);                      // wg_mask=0, data_size=1 (2 bytes)
  g1[1] = (int)((width_elems & 0xFFFFu) << 16);               // tensor_dim0 lo
  g1[2] = (int)(((width_elems >> 16) & 0xFFFFu) |
                ((rows & 0xFFFFu) << 16));                    // td0 hi | tensor_dim1 lo
  g1[3] = (int)(((rows >> 16) & 0xFFFFu) |
                ((width_elems & 0xFFFFu) << 16));             // td1 hi | tile_dim0
  g1[4] = (int)(rows & 0xFFFFu);                // tile_dim1 | tile_dim2=0
  g1[5] = (int)stride_elems;                    // tensor_dim0_stride[31:0]
  g1[6] = 0;                                    // stride0 hi | dim1_stride lo
  g1[7] = 0;
  tdm_v4i z4 = {0, 0, 0, 0};
#if defined(__clang_major__) && (__clang_major__ >= 23)
  tdm_v8i z8 = {0, 0, 0, 0, 0, 0, 0, 0};
  __builtin_amdgcn_tensor_load_to_lds(g0, g1, z4, z4, z8, 0);
#else
  __builtin_amdgcn_tensor_load_to_lds(g0, g1, z4, z4, 0);
#endif
}

// ---------------- Kernels ------------------------------------------------

// Pack weights transposed (N-major) + zero-pad to bf16.
__global__ void k_pack(const float* __restrict__ W, const float* __restrict__ ctx_w,
                       const float* __restrict__ gate_w,
                       bf16_t* __restrict__ WT, bf16_t* __restrict__ w1T,
                       bf16_t* __restrict__ w2T) {
  int idx = blockIdx.x * blockDim.x + threadIdx.x;
  if (idx < kDP * kDP) {                 // WT[e,d] = W[d,e]
    int e = idx / kDP, d = idx % kDP;
    float v = (e < kDim && d < kDim) ? W[(size_t)d * kDim + e] : 0.f;
    WT[idx] = (bf16_t)v;
  }
  if (idx < kNC * kK1) {                 // w1T[co, tap*DP+ci] = ctx_w[co,ci,tap]
    int co = idx / kK1, k = idx % kK1;
    int tap = k / kDP, ci = k % kDP;
    float v = (ci < kDim) ? ctx_w[((size_t)co * kDim + ci) * kKS + tap] : 0.f;
    w1T[idx] = (bf16_t)v;
  }
  if (idx < kNC * kK2) {                 // w2T[co, tap*GID+c], c<DP: ctx, else att
    int co = idx / kK2, k = idx % kK2;
    int tap = k / kGID, c = k % kGID;
    float v = 0.f;
    if (c < kDP) {
      if (c < kDim) v = gate_w[((size_t)co * 2 * kDim + c) * kKS + tap];
    } else {
      int ci = c - kDP;
      if (ci < kDim) v = gate_w[((size_t)co * 2 * kDim + kDim + ci) * kKS + tap];
    }
    w2T[idx] = (bf16_t)v;
  }
}

// Gather context embeddings with halo rows; also fill gate-input first half.
__global__ void k_gather_ctx(const int* __restrict__ ids,
                             const float* __restrict__ wordmat,
                             bf16_t* __restrict__ ctx_bf, bf16_t* __restrict__ gi) {
  int r = blockIdx.x;                    // b*kRows + row
  int b = r / kRows, row = r % kRows;
  int d = threadIdx.x;                   // 0..kDP-1
  float v = 0.f;
  if (row >= 1 && row <= kL1 && d < kDim) {
    int tok = ids[b * kL1 + row - 1];
    v = wordmat[(size_t)tok * kDim + d];
  }
  bf16_t bv = (bf16_t)v;
  ctx_bf[(size_t)r * kDP + d] = bv;
  gi[(size_t)r * kGID + d] = bv;
  if (row == 0 || row == kL1 + 1) gi[(size_t)r * kGID + kDP + d] = (bf16_t)0.f;
}

// Gather aspect embeddings: row-major (A operand for sy) and d-major (B for att).
__global__ void k_gather_asp(const int* __restrict__ ids,
                             const float* __restrict__ wordmat,
                             bf16_t* __restrict__ asp_bf, bf16_t* __restrict__ aspT) {
  int r = blockIdx.x;                    // b*kL2 + j
  int b = r / kL2, j = r % kL2;
  int d = threadIdx.x;
  float v = 0.f;
  if (d < kDim) {
    int tok = ids[b * kL2 + j];
    v = wordmat[(size_t)tok * kDim + d];
  }
  bf16_t bv = (bf16_t)v;
  asp_bf[(size_t)r * kDP + d] = bv;
  aspT[((size_t)b * kDP + d) * kL2 + j] = bv;
}

// sx = relu(context @ W): per batch M=512, N=320, K=320. 8 waves x (64x64).
__global__ void __launch_bounds__(256) k_gemm_sx(const bf16_t* __restrict__ ctx_bf,
                                                 const bf16_t* __restrict__ WT,
                                                 bf16_t* __restrict__ sx) {
  int lane = threadIdx.x & 31, w = threadIdx.x >> 5;
  int b = blockIdx.y, n_base = blockIdx.x * 64, m_base = w * 64;
  const bf16_t* A = ctx_bf + ((size_t)b * kRows + 1) * kDP;  // skip halo
  v8f acc[4][4];
  wave_gemm<4, 4>(A, kDP, WT, kDP, kDP, m_base, n_base, lane, acc);
  bf16_t* C = sx + (size_t)b * kL1 * kDP;
#pragma unroll
  for (int i = 0; i < 4; ++i)
#pragma unroll
    for (int j = 0; j < 4; ++j)
      store_bf16(C, kDP, m_base + i * 16, n_base + j * 16, lane, acc[i][j], true);
}

// sy = relu(aspect @ W): flat M = B*32 = 4096 rows.
__global__ void __launch_bounds__(256) k_gemm_sy(const bf16_t* __restrict__ asp_bf,
                                                 const bf16_t* __restrict__ WT,
                                                 bf16_t* __restrict__ sy) {
  int lane = threadIdx.x & 31, w = threadIdx.x >> 5;
  int n_base = blockIdx.x * 64, m_base = blockIdx.y * 512 + w * 64;
  v8f acc[4][4];
  wave_gemm<4, 4>(asp_bf, kDP, WT, kDP, kDP, m_base, n_base, lane, acc);
#pragma unroll
  for (int i = 0; i < 4; ++i)
#pragma unroll
    for (int j = 0; j < 4; ++j)
      store_bf16(sy, kDP, m_base + i * 16, n_base + j * 16, lane, acc[i][j], true);
}

// s[b] = sx[b] @ sy[b]^T : M=512, N=32, K=320 (sy row-major IS the BT operand).
__global__ void __launch_bounds__(256) k_gemm_s(const bf16_t* __restrict__ sx,
                                                const bf16_t* __restrict__ sy,
                                                float* __restrict__ s) {
  int lane = threadIdx.x & 31, w = threadIdx.x >> 5;
  int b = blockIdx.y, m_base = w * 64;
  const bf16_t* A  = sx + (size_t)b * kL1 * kDP;
  const bf16_t* BT = sy + (size_t)b * kL2 * kDP;
  v8f acc[4][2];
  wave_gemm<4, 2>(A, kDP, BT, kDP, kDP, m_base, 0, lane, acc);
  float* C = s + (size_t)b * kL1 * kL2;
#pragma unroll
  for (int i = 0; i < 4; ++i)
#pragma unroll
    for (int j = 0; j < 2; ++j)
      store_f32(C, kL2, m_base + i * 16, j * 16, lane, acc[i][j]);
}

// Softmax over dim L1 (per column j of each batch), masked by aspect mask.
__global__ void __launch_bounds__(512) k_softmax(const float* __restrict__ s,
                                                 const float* __restrict__ amask,
                                                 bf16_t* __restrict__ alpha) {
  int j = blockIdx.x, b = blockIdx.y, i = threadIdx.x;
  __shared__ float red[kL1];
  float v = s[((size_t)b * kL1 + i) * kL2 + j];
  red[i] = v;
  __syncthreads();
  for (int off = kL1 / 2; off > 0; off >>= 1) {
    if (i < off) red[i] = fmaxf(red[i], red[i + off]);
    __syncthreads();
  }
  float mx = red[0];
  __syncthreads();
  float e = expf(v - mx) * amask[b * kL2 + j];
  red[i] = e;
  __syncthreads();
  for (int off = kL1 / 2; off > 0; off >>= 1) {
    if (i < off) red[i] += red[i + off];
    __syncthreads();
  }
  float sum = red[0];
  alpha[((size_t)b * kL1 + i) * kL2 + j] = (bf16_t)(e / (sum + 1e-10f));
}

// att = alpha @ aspect; writes bf16 straight into gate-input second half.
__global__ void __launch_bounds__(256) k_gemm_att(const bf16_t* __restrict__ alpha,
                                                  const bf16_t* __restrict__ aspT,
                                                  bf16_t* __restrict__ gi) {
  int lane = threadIdx.x & 31, w = threadIdx.x >> 5;
  int b = blockIdx.y, n_base = blockIdx.x * 64, m_base = w * 64;
  const bf16_t* A  = alpha + (size_t)b * kL1 * kL2;   // lda=32, K=32
  const bf16_t* BT = aspT + (size_t)b * kDP * kL2;    // ldb=32
  v8f acc[4][4];
  wave_gemm<4, 4>(A, kL2, BT, kL2, kL2, m_base, n_base, lane, acc);
  bf16_t* C = gi + ((size_t)b * kRows + 1) * kGID + kDP;  // second half, skip halo
#pragma unroll
  for (int i = 0; i < 4; ++i)
#pragma unroll
    for (int j = 0; j < 4; ++j)
      store_bf16(C, kGID, m_base + i * 16, n_base + j * 16, lane, acc[i][j], false);
}

__global__ void k_init_pooled(float* __restrict__ pooled) {
  int i = blockIdx.x * blockDim.x + threadIdx.x;
  if (i < kB * kNC) pooled[i] = -1e30f;
}

// Fused dual conv GEMM (K=3 convs as overlapping-row GEMMs via halo layout)
// + tanh/relu gate + global max-pool over L.
// A tile (130 rows x 640 bf16 = 166KB) staged into LDS by the Tensor Data
// Mover; all A fragments then come from LDS, weights stream from L2.
// 8 waves = 4 row-subtiles (32) x 2 col-halves (32), dual accumulators.
__global__ void __launch_bounds__(256) k_gemm_conv(const bf16_t* __restrict__ gi,
                                                   const bf16_t* __restrict__ w1T,
                                                   const bf16_t* __restrict__ w2T,
                                                   const float* __restrict__ ctx_b,
                                                   const float* __restrict__ gate_b,
                                                   float* __restrict__ pooled) {
  extern __shared__ char smem_raw[];
  const bf16_t* As = (const bf16_t*)smem_raw;
  int lane = threadIdx.x & 31, w = threadIdx.x >> 5;
  int wm = w & 3, wn = w >> 2;
  int b = blockIdx.z;
  int n_w = blockIdx.x * 64 + wn * 32;
  int m_start = blockIdx.y * kConvRowsPerBlk;
  const bf16_t* Ag = gi + ((size_t)b * kRows + m_start) * kGID;
  if (w == 0) {
    tdm_load_2d_to_lds(Ag, 0u, (unsigned)kGID, (unsigned)kConvLdsRows,
                       (unsigned)kGID);
    __builtin_amdgcn_s_wait_tensorcnt(0);
  }
  __syncthreads();

  v8f accS[2][2], accG[2][2];
#pragma unroll
  for (int i = 0; i < 2; ++i)
#pragma unroll
    for (int j = 0; j < 2; ++j) {
      accS[i][j] = (v8f){0, 0, 0, 0, 0, 0, 0, 0};
      accG[i][j] = (v8f){0, 0, 0, 0, 0, 0, 0, 0};
    }
  for (int k = 0; k < kK2; k += 32) {
    int tap = k / kGID, km = k % kGID;
    bool inctx = km < kDP;                 // this K-chunk also feeds sentiment conv
    v16bf bg[2], bs[2];
#pragma unroll
    for (int j = 0; j < 2; ++j) bg[j] = load_b(w2T, kK2, n_w + j * 16, k, lane);
    if (inctx) {
      int ks = tap * kDP + km;
#pragma unroll
      for (int j = 0; j < 2; ++j) bs[j] = load_b(w1T, kK1, n_w + j * 16, ks, lane);
    }
#pragma unroll
    for (int i = 0; i < 2; ++i) {
      v16bf a = load_a(As, kGID, wm * 32 + i * 16, k, lane);  // from LDS
#pragma unroll
      for (int j = 0; j < 2; ++j) {
        accG[i][j] = wmma_bf16(a, bg[j], accG[i][j]);
        if (inctx) accS[i][j] = wmma_bf16(a, bs[j], accS[i][j]);
      }
    }
  }
  // Epilogue: tanh(sent+b1) * relu(gate+b2), max over this wave's 32 rows.
#pragma unroll
  for (int j = 0; j < 2; ++j) {
    int co = n_w + j * 16 + (lane & 15);
    float b1v = ctx_b[co], b2v = gate_b[co];
    float mx = -1e30f;
#pragma unroll
    for (int i = 0; i < 2; ++i)
#pragma unroll
      for (int r = 0; r < 8; ++r) {
        float sv = tanhf(accS[i][j][r] + b1v);
        float gv = accG[i][j][r] + b2v;
        gv = gv > 0.f ? gv : 0.f;
        mx = fmaxf(mx, sv * gv);
      }
    atomicMaxF(&pooled[(size_t)b * kNC + co], mx);
  }
}

// logits = pooled @ lin_w^T + lin_b  (128x256 @ 256x3)
__global__ void k_final(const float* __restrict__ pooled,
                        const float* __restrict__ lin_w,
                        const float* __restrict__ lin_b, float* __restrict__ out) {
  int idx = blockIdx.x * blockDim.x + threadIdx.x;
  if (idx >= kB * kNCls) return;
  int b = idx / kNCls, c = idx % kNCls;
  float acc = lin_b[c];
  for (int co = 0; co < kNC; ++co)
    acc += pooled[(size_t)b * kNC + co] * lin_w[(size_t)c * kNC + co];
  out[idx] = acc;
}

}  // namespace

extern "C" void kernel_launch(void* const* d_in, const int* in_sizes, int n_in,
                              void* d_out, int out_size, void* d_ws, size_t ws_size,
                              hipStream_t stream) {
  const int*   context_ids = (const int*)d_in[0];
  // d_in[1] context_masks: unused by the reference
  const int*   aspect_ids  = (const int*)d_in[2];
  const float* aspect_mask = (const float*)d_in[3];
  const float* wordmat     = (const float*)d_in[4];
  const float* W           = (const float*)d_in[5];
  const float* ctx_w       = (const float*)d_in[6];
  const float* ctx_b       = (const float*)d_in[7];
  const float* gate_w      = (const float*)d_in[8];
  const float* gate_b      = (const float*)d_in[9];
  const float* lin_w       = (const float*)d_in[10];
  const float* lin_b       = (const float*)d_in[11];
  float* out = (float*)d_out;

  // Workspace carving (256B aligned).
  size_t off = 0;
  auto carve = [&](size_t bytes) {
    void* p = (char*)d_ws + off;
    off += (bytes + 255) & ~(size_t)255;
    return p;
  };
  bf16_t* ctx_bf = (bf16_t*)carve((size_t)kB * kRows * kDP * sizeof(bf16_t));
  bf16_t* gi     = (bf16_t*)carve((size_t)kB * kRows * kGID * sizeof(bf16_t));
  bf16_t* asp_bf = (bf16_t*)carve((size_t)kB * kL2 * kDP * sizeof(bf16_t));
  bf16_t* aspT   = (bf16_t*)carve((size_t)kB * kDP * kL2 * sizeof(bf16_t));
  bf16_t* WT     = (bf16_t*)carve((size_t)kDP * kDP * sizeof(bf16_t));
  bf16_t* w1T    = (bf16_t*)carve((size_t)kNC * kK1 * sizeof(bf16_t));
  bf16_t* w2T    = (bf16_t*)carve((size_t)kNC * kK2 * sizeof(bf16_t));
  bf16_t* sx     = (bf16_t*)carve((size_t)kB * kL1 * kDP * sizeof(bf16_t));
  bf16_t* sy     = (bf16_t*)carve((size_t)kB * kL2 * kDP * sizeof(bf16_t));
  float*  s      = (float*)carve((size_t)kB * kL1 * kL2 * sizeof(float));
  bf16_t* alpha  = (bf16_t*)carve((size_t)kB * kL1 * kL2 * sizeof(bf16_t));
  float*  pooled = (float*)carve((size_t)kB * kNC * sizeof(float));
  (void)ws_size; (void)n_in; (void)in_sizes; (void)out_size;

  // 1. Pack weights (transposed, bf16, padded).
  k_pack<<<(kNC * kK2 + 255) / 256, 256, 0, stream>>>(W, ctx_w, gate_w, WT, w1T, w2T);
  // 2. Embedding gathers.
  k_gather_ctx<<<kB * kRows, kDP, 0, stream>>>(context_ids, wordmat, ctx_bf, gi);
  k_gather_asp<<<kB * kL2, kDP, 0, stream>>>(aspect_ids, wordmat, asp_bf, aspT);
  k_init_pooled<<<(kB * kNC + 255) / 256, 256, 0, stream>>>(pooled);
  // 3. sx = relu(ctx@W), sy = relu(asp@W).
  k_gemm_sx<<<dim3(kDP / 64, kB), 256, 0, stream>>>(ctx_bf, WT, sx);
  k_gemm_sy<<<dim3(kDP / 64, (kB * kL2) / 512), 256, 0, stream>>>(asp_bf, WT, sy);
  // 4. s = sx @ sy^T per batch.
  k_gemm_s<<<dim3(1, kB), 256, 0, stream>>>(sx, sy, s);
  // 5. Softmax over L1 per (b, j).
  k_softmax<<<dim3(kL2, kB), kL1, 0, stream>>>(s, aspect_mask, alpha);
  // 6. att = alpha @ aspect -> gate-input second half.
  k_gemm_att<<<dim3(kDP / 64, kB), 256, 0, stream>>>(alpha, aspT, gi);
  // 7. Fused dual conv GEMM (TDM-staged LDS A tile) + tanh*relu + max-pool.
  k_gemm_conv<<<dim3(kNC / 64, kL1 / kConvRowsPerBlk, kB), 256, kConvLdsBytes,
                stream>>>(gi, w1T, w2T, ctx_b, gate_b, pooled);
  // 8. Final linear.
  k_final<<<(kB * kNCls + 255) / 256, 256, 0, stream>>>(pooled, lin_w, lin_b, out);
}
